// CTRGC_39676907888771
// MI455X (gfx1250) — matrix operand order repaired and checked by reading, not compiled, not run on hardware
//
#include <hip/hip_runtime.h>
#include <hip/hip_bf16.h>

typedef __attribute__((ext_vector_type(16))) __bf16 v16bf;
typedef __attribute__((ext_vector_type(8)))  float  v8f;

#define NB     64
#define C_IN   128
#define C_OUT  128
#define T_DIM  128
#define V_DIM  25
#define REL    16

#define XTILE_ELEMS (25 * 4 * 32 * 16)   // bf16 elems per (n, t-tile) packed block = 51200
#define W3_ELEMS    (8 * 4 * 32 * 16)    // bf16 elems of packed W3 = 16384

// ---------------------------------------------------------------------------
// Kernel 1: xm = x.mean(T)  AND  pack x into bf16 WMMA B-fragment layout.
// Reads every element of x exactly once (105 MB); packed copy (52 MB) stays
// dirty in L2 for the async-LDS fill in kernel 3.
// Thread map: lane = v, group g = t-tile (g==tt since T tiles are 16 wide).
// ---------------------------------------------------------------------------
__global__ void mean_pack_kernel(const float* __restrict__ x,
                                 float* __restrict__ xm,
                                 __bf16* __restrict__ xbf) {
  const int nc = blockIdx.x;          // n*C_IN + c
  const int n  = nc >> 7;
  const int c  = nc & 127;
  const int v  = threadIdx.x & 31;
  const int g  = threadIdx.x >> 5;    // 0..7 == t-tile
  __shared__ float part[8][32];
  float s = 0.f;
  if (v < V_DIM) {
    const float* p = x + ((size_t)nc * T_DIM) * V_DIM + v;
    const int ks = c >> 5, k = c & 31;
    const int lane2b = ((k >> 3) & 1) ? 16 : 0;   // B-frag: k octets 0,2 -> lanes 0-15; 1,3 -> lanes 16-31
    const int e = ((k >> 4) << 3) + (k & 7);
    __bf16* dst = xbf + (size_t)(n * 8 + g) * XTILE_ELEMS;
    for (int i = 0; i < 16; ++i) {
      const float val = p[(g * 16 + i) * V_DIM];
      s += val;
      const int col = i * V_DIM + v;        // col within (16 x 25) tile
      const int ct = col >> 4, nb = col & 15;
      dst[(((ct * 4 + ks) * 32) + (lane2b + nb)) * 16 + e] = (__bf16)val;
    }
  }
  part[g][v] = s;
  __syncthreads();
  if (g == 0 && v < V_DIM) {
    float acc = 0.f;
    for (int i = 0; i < 8; ++i) acc += part[i][v];
    xm[nc * V_DIM + v] = acc * (1.0f / T_DIM);
  }
}

// ---------------------------------------------------------------------------
// Kernel 1b: pack W3 (128x128) into the 16-bit A-matrix fragment layout once.
// ---------------------------------------------------------------------------
__global__ void pack_w3_kernel(const float* __restrict__ W3,
                               __bf16* __restrict__ w3bf) {
  for (int idx = threadIdx.x; idx < C_OUT * C_IN; idx += blockDim.x) {
    const int o = idx >> 7, cc = idx & 127;
    const int ot = o >> 4, m = o & 15;
    const int ks = cc >> 5, k = cc & 31;
    const int oct = k >> 3, r8 = k & 7;
    const int lane2 = m + ((oct & 1) ? 16 : 0);
    const int e = 2 * (((oct >> 1) ? 4 : 0) + (r8 >> 1)) + (k & 1);
    w3bf[(((ot * 4 + ks) * 32) + lane2) * 16 + e] = (__bf16)W3[idx];
  }
}

// ---------------------------------------------------------------------------
// Kernel 2: per-n: x1/x2 (R x V), d = tanh(x1-x2), adj = alpha*(W4 d + b4) + A
// ---------------------------------------------------------------------------
__global__ void adj_kernel(const float* __restrict__ xm,
                           const float* __restrict__ A,
                           const float* __restrict__ alpha_p,
                           const float* __restrict__ W1, const float* __restrict__ b1,
                           const float* __restrict__ W2, const float* __restrict__ b2,
                           const float* __restrict__ W4, const float* __restrict__ b4,
                           float* __restrict__ adj) {
  const int n   = blockIdx.x;
  const int tid = threadIdx.x;
  __shared__ float xm_s[C_IN * V_DIM];
  __shared__ float x1_s[REL * V_DIM];
  __shared__ float x2_s[REL * V_DIM];
  __shared__ float d_s[REL * V_DIM * V_DIM];

  for (int i = tid; i < C_IN * V_DIM; i += blockDim.x)
    xm_s[i] = xm[n * C_IN * V_DIM + i];
  __syncthreads();

  for (int j = tid; j < REL * V_DIM; j += blockDim.x) {
    const int r = j / V_DIM, v = j % V_DIM;
    float s1 = b1[r], s2 = b2[r];
    for (int c = 0; c < C_IN; ++c) {
      const float xv = xm_s[c * V_DIM + v];
      s1 += W1[r * C_IN + c] * xv;
      s2 += W2[r * C_IN + c] * xv;
    }
    x1_s[j] = s1; x2_s[j] = s2;
  }
  __syncthreads();

  for (int j = tid; j < REL * V_DIM * V_DIM; j += blockDim.x) {
    const int r   = j / (V_DIM * V_DIM);
    const int rem = j % (V_DIM * V_DIM);
    const int v = rem / V_DIM, u = rem % V_DIM;
    d_s[j] = tanhf(x1_s[r * V_DIM + v] - x2_s[r * V_DIM + u]);
  }
  __syncthreads();

  const float alpha = alpha_p[0];
  for (int j = tid; j < C_OUT * V_DIM * V_DIM; j += blockDim.x) {
    const int o   = j / (V_DIM * V_DIM);
    const int rem = j % (V_DIM * V_DIM);
    float s = b4[o];
    for (int r = 0; r < REL; ++r)
      s += W4[o * REL + r] * d_s[r * V_DIM * V_DIM + rem];
    adj[(size_t)n * C_OUT * V_DIM * V_DIM + j] = s * alpha + A[rem];
  }
}

// ---------------------------------------------------------------------------
// Kernel 3: async-LDS fill of pre-packed fragments, then
//   Stage A: x3 = W3 @ x_tile (bf16 WMMA, K=128)
//   Stage B: out = x3 @ adj   (bf16 WMMA, K=V pad 32)
// ---------------------------------------------------------------------------
#define LDS_A_OFF   0          // W3 frags  32 KB
#define LDS_B_OFF   32768      // x  frags 100 KB
#define LDS_X3_OFF  135168     // x3 frags 128 KB
#define LDS_TOTAL   266240     // 260 KB of 320 KB

__device__ __forceinline__ void async_lds_b128(uint32_t lds_off, uint64_t gaddr) {
  asm volatile("global_load_async_to_lds_b128 %0, %1, off"
               :: "v"(lds_off), "v"(gaddr) : "memory");
}

__global__ void fused_gemm_kernel(const __bf16* __restrict__ xbf,
                                  const __bf16* __restrict__ w3bf,
                                  const float* __restrict__ b3,
                                  const float* __restrict__ adj,
                                  float* __restrict__ out) {
  extern __shared__ char smem[];
  __bf16* aL  = (__bf16*)(smem + LDS_A_OFF);
  __bf16* bL  = (__bf16*)(smem + LDS_B_OFF);
  __bf16* x3L = (__bf16*)(smem + LDS_X3_OFF);

  const int tt   = blockIdx.x;       // t-tile 0..7
  const int n    = blockIdx.y;       // batch
  const int t0   = tt * 16;
  const int tid  = threadIdx.x;
  const int lane = tid & 31;
  const int wid  = tid >> 5;

  // --- Async DMA: fill W3 fragments (32 KB) and x fragments (100 KB). ---
  {
    const uint64_t gw = (uint64_t)(uintptr_t)w3bf;
    const uint32_t la = (uint32_t)(uintptr_t)(smem + LDS_A_OFF);
    for (int i = tid; i < (W3_ELEMS * 2) / 16; i += 256)       // 2048 x b128
      async_lds_b128(la + i * 16, gw + (uint64_t)i * 16);

    const uint64_t gx = (uint64_t)(uintptr_t)(xbf + (size_t)(n * 8 + tt) * XTILE_ELEMS);
    const uint32_t lb = (uint32_t)(uintptr_t)(smem + LDS_B_OFF);
    for (int i = tid; i < (XTILE_ELEMS * 2) / 16; i += 256)    // 6400 x b128
      async_lds_b128(lb + i * 16, gx + (uint64_t)i * 16);
  }

  // Zero the x3 region (K=25..31 zero padding for stage B) while DMA runs.
  {
    uint32_t* p = (uint32_t*)x3L;
    for (int i = tid; i < (128 * 32 * 16 * 2) / 4; i += 256) p[i] = 0u;
  }

  asm volatile("s_wait_asynccnt 0x0" ::: "memory");
  __syncthreads();

  // ---- Stage A: x3_tile = W3 @ x_tile.  8 o-tiles x 25 col-tiles, K=128. ----
  for (int tix = wid; tix < 200; tix += 8) {
    const int ot = tix / 25, ct = tix % 25;
    v8f acc = {};
    for (int ks = 0; ks < 4; ++ks) {
      v16bf av = *(const v16bf*)(aL + (((ot * 4 + ks) * 32) + lane) * 16);
      v16bf bv = *(const v16bf*)(bL + (((ct * 4 + ks) * 32) + lane) * 16);
      acc = __builtin_amdgcn_wmma_f32_16x16x32_bf16(false, av, false, bv,
                                                    (short)0, acc, false, false);
    }
    // Scatter D (rows=o, col=(t,v)) into x3 A-fragment layout (rows=t, K=v), +b3.
    const int hi  = (lane < 16) ? 0 : 8;
    const int col = ct * 16 + (lane & 15);
    const int t = col / V_DIM, v = col % V_DIM;
    const int oct = v >> 3, r8 = v & 7;
    const int lane2 = t + ((oct & 1) ? 16 : 0);
    const int e = 2 * (((oct >> 1) ? 4 : 0) + (r8 >> 1)) + (v & 1);
    for (int i = 0; i < 8; ++i) {
      const int ch = ot * 16 + i + hi;
      x3L[(ch * 32 + lane2) * 16 + e] = (__bf16)(acc[i] + b3[ch]);
    }
  }
  __syncthreads();

  // ---- Stage B: out[n,ch,t0:t0+16,:] = x3[ch] @ adj[n,ch]  (K=25 pad 32). ----
  const float* adjn = adj + (size_t)n * C_OUT * V_DIM * V_DIM;
  for (int job = wid; job < 256; job += 8) {
    const int ch = job >> 1, ut = job & 1;
    v16bf av = *(const v16bf*)(x3L + (ch * 32 + lane) * 16);
    const int u = ut * 16 + (lane & 15);
    const int kbase = (lane < 16) ? 0 : 8;
    const float* ap = adjn + ch * V_DIM * V_DIM;
    v16bf bv;
    for (int e = 0; e < 16; ++e) {
      const int v = kbase + (e & 7) + ((e & 8) << 1);
      const float val = (v < V_DIM && u < V_DIM) ? ap[v * V_DIM + u] : 0.f;
      bv[e] = (__bf16)val;
    }
    v8f acc = {};
    acc = __builtin_amdgcn_wmma_f32_16x16x32_bf16(false, av, false, bv,
                                                  (short)0, acc, false, false);
    if (u < V_DIM) {
      const int mbase = (lane < 16) ? 0 : 8;
      for (int i = 0; i < 8; ++i) {
        const int t = t0 + i + mbase;
        out[(((size_t)n * C_OUT + ch) * T_DIM + t) * V_DIM + u] = acc[i];
      }
    }
  }
}

// ---------------------------------------------------------------------------
extern "C" void kernel_launch(void* const* d_in, const int* in_sizes, int n_in,
                              void* d_out, int out_size, void* d_ws, size_t ws_size,
                              hipStream_t stream) {
  const float* x     = (const float*)d_in[0];
  const float* A     = (const float*)d_in[1];
  const float* alpha = (const float*)d_in[2];
  const float* W1    = (const float*)d_in[3];
  const float* b1    = (const float*)d_in[4];
  const float* W2    = (const float*)d_in[5];
  const float* b2    = (const float*)d_in[6];
  const float* W3    = (const float*)d_in[7];
  const float* b3    = (const float*)d_in[8];
  const float* W4    = (const float*)d_in[9];
  const float* b4    = (const float*)d_in[10];
  float* out = (float*)d_out;

  // Scratch layout (bytes):
  //   xm   @ 0          :   819,200  (N*C_IN*V f32)
  //   adj  @ 819,200    : 20,480,000 (N*C_OUT*V*V f32)
  //   w3bf @ 21,299,200 :     32,768 (packed W3 bf16 frags)
  //   xbf  @ 21,331,968 : 52,428,800 (packed x bf16 frags)   total ~70.3 MB
  float*  xm   = (float*)d_ws;
  float*  adj  = xm + NB * C_IN * V_DIM;
  __bf16* w3bf = (__bf16*)((char*)d_ws + 21299200);
  __bf16* xbf  = (__bf16*)((char*)d_ws + 21331968);

  mean_pack_kernel<<<dim3(NB * C_IN), dim3(256), 0, stream>>>(x, xm, xbf);
  pack_w3_kernel<<<dim3(1), dim3(256), 0, stream>>>(W3, w3bf);
  adj_kernel<<<dim3(NB), dim3(256), 0, stream>>>(xm, A, alpha, W1, b1, W2, b2, W4, b4, adj);
  fused_gemm_kernel<<<dim3(T_DIM / 16, NB), dim3(256), LDS_TOTAL, stream>>>(xbf, w3bf, b3, adj, out);
}
